// LSTMPPOPolicy_57415122813289
// MI455X (gfx1250) — compile-verified, matrix operand order checked by reading, not run on hardware
//
#include <hip/hip_runtime.h>
#include <cstddef>
#include <cstdint>

// ---------------------------------------------------------------------------
// Types for CDNA5 WMMA (wave32, v_wmma_f32_16x16x32_bf16)
// ---------------------------------------------------------------------------
typedef __attribute__((ext_vector_type(16))) __bf16 v16bf;
typedef __attribute__((ext_vector_type(8)))  __bf16 v8bf;
typedef __attribute__((ext_vector_type(8)))  float  v8f;
typedef __attribute__((ext_vector_type(4)))  unsigned int u32x4;

static constexpr int Bz = 32, Tz = 512, OBSz = 256, ENCz = 1024, Hz = 1024,
                     FHz = 4096, Az = 64;

#define DEV __device__ __forceinline__

DEV unsigned short f2bf(float f) {
  unsigned int u = __float_as_uint(f);
  unsigned int r = u + 0x7FFFu + ((u >> 16) & 1u);  // round-to-nearest-even
  return (unsigned short)(r >> 16);
}
DEV float sigmoid_f(float x) { return 1.0f / (1.0f + __expf(-x)); }

DEV v8f v8f_zero() {
  v8f z;
#pragma unroll
  for (int i = 0; i < 8; ++i) z[i] = 0.0f;
  return z;
}

union bfpack { u32x4 u; v8bf b; };

// Load a 16x32 bf16 fragment from GLOBAL (A-layout; B mirrored, W is (N,K)
// row-major => columns of B are W rows).
DEV v16bf load_frag(const unsigned short* __restrict__ base, int ld,
                    int rowbase, int k0, int lane) {
  const int half = lane >> 4;
  const unsigned short* p =
      base + (size_t)(rowbase + (lane & 15)) * (size_t)ld + k0 + half * 8;
  bfpack lo, hi;
  lo.u = *reinterpret_cast<const u32x4*>(p);        // global_load_b128
  hi.u = *reinterpret_cast<const u32x4*>(p + 16);   // global_load_b128
  v16bf v;
#pragma unroll
  for (int i = 0; i < 8; ++i) { v[i] = lo.b[i]; v[8 + i] = hi.b[i]; }
  return v;
}

// Same fragment gather from LDS (ds_load_b128).  Row stride padded to spread
// banks: lanes 0-15 tile banks 4r..4r+3 -> all 64 banks once.
#define WLDS_LD 1032  // 1024 + 8 bf16 elements (16B) padding
DEV v16bf load_frag_lds(const unsigned short* base, int rowbase, int k0,
                        int lane) {
  const int half = lane >> 4;
  const unsigned short* p =
      base + (rowbase + (lane & 15)) * WLDS_LD + k0 + half * 8;
  bfpack lo, hi;
  lo.u = *reinterpret_cast<const u32x4*>(p);        // ds_load_b128
  hi.u = *reinterpret_cast<const u32x4*>(p + 16);   // ds_load_b128
  v16bf v;
#pragma unroll
  for (int i = 0; i < 8; ++i) { v[i] = lo.b[i]; v[8 + i] = hi.b[i]; }
  return v;
}

DEV v8f wmma_bf16(v16bf a, v16bf b, v8f c) {
  return __builtin_amdgcn_wmma_f32_16x16x32_bf16(false, a, false, b,
                                                 (short)0, c, false, false);
}

// ---------------------------------------------------------------------------
// Generic WMMA GEMM: C[M,N] = act(A[M,K] @ W[N,K]^T + bias)
// Wave tile: 32(M) x 64(N).  Requires M%32==0, N%64==0, K%32==0.
// ---------------------------------------------------------------------------
__global__ __launch_bounds__(256) void wmma_gemm_kernel(
    const unsigned short* __restrict__ A, const unsigned short* __restrict__ W,
    const float* __restrict__ bias, float* __restrict__ Cf,
    unsigned short* __restrict__ Cb, int M, int N, int K, int act) {
  const int wave = (int)((blockIdx.x * blockDim.x + threadIdx.x) >> 5);
  const int lane = (int)(threadIdx.x & 31);
  const int NT = N >> 6;
  const int MT = M >> 5;
  if (wave >= MT * NT) return;  // whole-wave uniform exit
  const int mt = wave / NT, nt = wave % NT;
  const int mbase = mt * 32, nbase = nt * 64;

  v8f acc[2][4];
#pragma unroll
  for (int i = 0; i < 2; ++i)
#pragma unroll
    for (int j = 0; j < 4; ++j) acc[i][j] = v8f_zero();

  for (int k0 = 0; k0 < K; k0 += 32) {
    __builtin_prefetch(A + (size_t)(mbase + (lane & 15)) * K + k0 + 32, 0, 1);
    v16bf a0 = load_frag(A, K, mbase, k0, lane);
    v16bf a1 = load_frag(A, K, mbase + 16, k0, lane);
#pragma unroll
    for (int j = 0; j < 4; ++j) {
      v16bf b = load_frag(W, K, nbase + j * 16, k0, lane);
      acc[0][j] = wmma_bf16(a0, b, acc[0][j]);
      acc[1][j] = wmma_bf16(a1, b, acc[1][j]);
    }
  }

  // Epilogue.  C/D fragment: elem r -> M = 16*i + r + 8*(lane>=16), N = lane%16
  const int half = lane >> 4, col = lane & 15;
#pragma unroll
  for (int i = 0; i < 2; ++i) {
#pragma unroll
    for (int j = 0; j < 4; ++j) {
      const int n = nbase + j * 16 + col;
      const float bval = bias ? bias[n] : 0.0f;
#pragma unroll
      for (int r = 0; r < 8; ++r) {
        const int m = mbase + i * 16 + r + half * 8;
        float x = acc[i][j][r] + bval;
        if (act) x = x * sigmoid_f(x);  // SiLU
        const size_t idx = (size_t)m * (size_t)N + (size_t)n;
        if (Cf) Cf[idx] = x;
        if (Cb) Cb[idx] = f2bf(x);
      }
    }
  }
}

// ---------------------------------------------------------------------------
// Persistent cooperative LSTM.  One launch, t-loop inside, grid barrier
// between steps (agent-scope atomic + fences).
//
// 64 blocks x 256 threads.  Block b owns hidden cols [16b, 16b+16) for all 4
// gates and all 32 batch rows.  The first half (K < 512, 64 KB) of its 64-row W_hh slice
// is preloaded into LDS once and served by ds_load_b128 for all 512 steps.
// The 8 waves split K=1024 into 128-wide partials (32 WMMAs on the critical
// path instead of 256), reduce through LDS, then 512 (m,col) owners apply
// the gate math.  Dynamic LDS: 64 KB partials + ~129 KB weights (< 320 KB).
// ---------------------------------------------------------------------------
#define LSTM_BLOCKS 64
#define PART_FLOATS (8 * 2 * 4 * 8 * 32)              // 16384 f32 = 64 KB
#define WLDS_SHORTS (64 * WLDS_LD)                    // 66048 bf16 = 129 KB
#define LSTM_SMEM_BYTES (PART_FLOATS * 4 + WLDS_SHORTS * 2)

// part[w][i][g][r][lane] : conflict-free (lane fastest)
DEV int part_idx(int w, int i, int g, int r, int lane) {
  return ((((w * 2 + i) * 4 + g) * 8 + r) * 32) + lane;
}

__global__ __launch_bounds__(256) void lstm_persistent_kernel(
    const float* __restrict__ gx,            // (B,T,4H) f32
    const unsigned short* __restrict__ Whh,  // (4H,H) bf16
    unsigned short* __restrict__ hping,      // (B,H) bf16  (t even reads)
    unsigned short* __restrict__ hpong,      // (B,H) bf16
    float* __restrict__ c,                   // (B,H) f32 (in place)
    float* __restrict__ out,                 // (B,T,H) f32
    float* __restrict__ hf,                  // (B,H) f32 (final h)
    unsigned int* __restrict__ ctr) {
  extern __shared__ char smem_raw[];
  float* part = (float*)smem_raw;                              // 64 KB
  unsigned short* wlds = (unsigned short*)(smem_raw + PART_FLOATS * 4);

  const int tid = (int)threadIdx.x;
  const int wave = tid >> 5, lane = tid & 31;
  const int hbase = (int)blockIdx.x * 16;
  const int kbase = wave * 128;

  // ---- one-time preload of this block's W_hh slice into LDS ----
  // local row j in [0,64): gate g = j/16, hidden col = hbase + j%16.
  for (int idx = tid; idx < 64 * 64; idx += 256) {
    const int j = idx >> 6;   // local row
    const int ch = idx & 63;  // 16B chunk along K
    const int grow = (j >> 4) * Hz + hbase + (j & 15);
    *reinterpret_cast<u32x4*>(wlds + j * WLDS_LD + ch * 8) =
        *reinterpret_cast<const u32x4*>(Whh + (size_t)grow * Hz + ch * 8);
  }
  __syncthreads();

  for (int t = 0; t < Tz; ++t) {
    const unsigned short* hin = (t & 1) ? hpong : hping;
    unsigned short* hout = (t & 1) ? hping : hpong;

    // -- K-partial WMMA accumulation (K in [kbase, kbase+128)) --
    v8f acc[2][4];
#pragma unroll
    for (int i = 0; i < 2; ++i)
#pragma unroll
      for (int g = 0; g < 4; ++g) acc[i][g] = v8f_zero();

#pragma unroll
    for (int k0 = 0; k0 < 128; k0 += 32) {
      v16bf a0 = load_frag(hin, Hz, 0, kbase + k0, lane);
      v16bf a1 = load_frag(hin, Hz, 16, kbase + k0, lane);
#pragma unroll
      for (int g = 0; g < 4; ++g) {
        v16bf b = load_frag_lds(wlds, g * 16, kbase + k0, lane);
        acc[0][g] = wmma_bf16(a0, b, acc[0][g]);
        acc[1][g] = wmma_bf16(a1, b, acc[1][g]);
      }
    }

    // -- stash partials in LDS --
#pragma unroll
    for (int i = 0; i < 2; ++i)
#pragma unroll
      for (int g = 0; g < 4; ++g)
#pragma unroll
        for (int r = 0; r < 8; ++r)
          part[part_idx(wave, i, g, r, lane)] = acc[i][g][r];
    __syncthreads();

    // -- reduce 8 partials + gate nonlinearities; 512 owners, 2 per thread --
    for (int cc = tid; cc < 512; cc += 256) {
      const int i = cc >> 8;
      const int rem = cc & 255;
      const int r = rem >> 5;
      const int ln = rem & 31;
      float gsum[4];
#pragma unroll
      for (int g = 0; g < 4; ++g) {
        float s = 0.0f;
#pragma unroll
        for (int w = 0; w < 8; ++w) s += part[part_idx(w, i, g, r, ln)];
        gsum[g] = s;
      }
      const int m = i * 16 + r + 8 * (ln >> 4);       // batch row
      const int col = hbase + (ln & 15);              // hidden col
      const float* gxp = gx + ((size_t)m * Tz + (size_t)t) * (size_t)FHz;
      const float gi = gsum[0] + gxp[0 * Hz + col];
      const float gf = gsum[1] + gxp[1 * Hz + col];
      const float gg = gsum[2] + gxp[2 * Hz + col];
      const float go = gsum[3] + gxp[3 * Hz + col];
      const size_t ci = (size_t)m * Hz + (size_t)col;
      const float cn = sigmoid_f(gf) * c[ci] + sigmoid_f(gi) * tanhf(gg);
      c[ci] = cn;
      const float hn = sigmoid_f(go) * tanhf(cn);
      out[((size_t)m * Tz + (size_t)t) * (size_t)Hz + (size_t)col] = hn;
      hout[ci] = f2bf(hn);
      if (t == Tz - 1) hf[ci] = hn;
    }

    // -- grid barrier: all 64 blocks must finish step t before t+1 --
    __syncthreads();
    if (tid == 0) {
      __builtin_amdgcn_fence(__ATOMIC_RELEASE, "agent");
      __hip_atomic_fetch_add(ctr, 1u, __ATOMIC_RELAXED,
                             __HIP_MEMORY_SCOPE_AGENT);
      const unsigned int target = (unsigned int)LSTM_BLOCKS * (unsigned)(t + 1);
      while (__hip_atomic_load(ctr, __ATOMIC_RELAXED,
                               __HIP_MEMORY_SCOPE_AGENT) < target) {
        __builtin_amdgcn_s_sleep(2);
      }
      __builtin_amdgcn_fence(__ATOMIC_ACQUIRE, "agent");
    }
    __syncthreads();
  }
}

// ---------------------------------------------------------------------------
// LayerNorm over H per (b,t) row; writes f32 in place + bf16 copy for heads.
// ---------------------------------------------------------------------------
__global__ __launch_bounds__(256) void ln_kernel(
    float* __restrict__ out, unsigned short* __restrict__ outb,
    const float* __restrict__ g, const float* __restrict__ b) {
  float* p = out + (size_t)blockIdx.x * Hz;
  unsigned short* pb = outb + (size_t)blockIdx.x * Hz;
  float s = 0.0f, s2 = 0.0f;
  for (int i = threadIdx.x; i < Hz; i += 256) {
    const float x = p[i]; s += x; s2 += x * x;
  }
#pragma unroll
  for (int o = 16; o; o >>= 1) {
    s += __shfl_xor(s, o, 32); s2 += __shfl_xor(s2, o, 32);
  }
  __shared__ float sh1[8], sh2[8];
  const int wid = threadIdx.x >> 5, lane = threadIdx.x & 31;
  if (lane == 0) { sh1[wid] = s; sh2[wid] = s2; }
  __syncthreads();
  float S = 0.0f, S2 = 0.0f;
#pragma unroll
  for (int w = 0; w < 8; ++w) { S += sh1[w]; S2 += sh2[w]; }
  const float mu = S / (float)Hz;
  const float var = S2 / (float)Hz - mu * mu;
  const float inv = rsqrtf(var + 1e-5f);
  for (int i = threadIdx.x; i < Hz; i += 256) {
    const float x = (p[i] - mu) * inv * g[i] + b[i];
    p[i] = x; pb[i] = f2bf(x);
  }
}

// ---------------------------------------------------------------------------
// ar/tar losses: grid-stride reduce + atomic into pre-zeroed d_out slots.
// ---------------------------------------------------------------------------
__global__ void zero_scalars_kernel(float* a, float* b, unsigned int* ctr) {
  if (threadIdx.x == 0 && blockIdx.x == 0) { *a = 0.0f; *b = 0.0f; *ctr = 0u; }
}

__global__ __launch_bounds__(256) void loss_kernel(
    const float* __restrict__ out, float* __restrict__ ar_dst,
    float* __restrict__ tar_dst) {
  const size_t N = (size_t)Bz * Tz * Hz;
  float ar = 0.0f, tar = 0.0f;
  for (size_t i = (size_t)blockIdx.x * 256 + threadIdx.x; i < N;
       i += (size_t)gridDim.x * 256) {
    const float x = out[i];
    ar += x * x;
    const size_t t = (i / (size_t)Hz) % (size_t)Tz;
    if (t) { const float d = x - out[i - Hz]; tar += d * d; }
  }
#pragma unroll
  for (int o = 16; o; o >>= 1) {
    ar += __shfl_xor(ar, o, 32); tar += __shfl_xor(tar, o, 32);
  }
  __shared__ float sa[8], st[8];
  const int wid = threadIdx.x >> 5, lane = threadIdx.x & 31;
  if (lane == 0) { sa[wid] = ar; st[wid] = tar; }
  __syncthreads();
  if (threadIdx.x == 0) {
    float A1 = 0.0f, A2 = 0.0f;
    for (int w = 0; w < 8; ++w) { A1 += sa[w]; A2 += st[w]; }
    const float ars = 0.01f / (float)((size_t)Bz * Tz * Hz);
    const float trs = 0.01f / (float)((size_t)Bz * (Tz - 1) * Hz);
    atomicAdd(ar_dst, A1 * ars);
    atomicAdd(tar_dst, A2 * trs);
  }
}

// ---------------------------------------------------------------------------
// Value head: one wave per (b,t) row, dot(out_row, Wc) + bc.
// ---------------------------------------------------------------------------
__global__ __launch_bounds__(256) void values_kernel(
    const float* __restrict__ out, const float* __restrict__ Wc,
    const float* __restrict__ bc, float* __restrict__ dst) {
  const int row = (int)((blockIdx.x * blockDim.x + threadIdx.x) >> 5);
  const int lane = (int)(threadIdx.x & 31);
  if (row >= Bz * Tz) return;
  const float* p = out + (size_t)row * Hz;
  float s = 0.0f;
  for (int i = lane; i < Hz; i += 32) s += p[i] * Wc[i];
#pragma unroll
  for (int o = 16; o; o >>= 1) s += __shfl_xor(s, o, 32);
  if (lane == 0) dst[row] = s + bc[0];
}

// ---------------------------------------------------------------------------
// Small elementwise helpers.
// ---------------------------------------------------------------------------
__global__ __launch_bounds__(256) void f32_to_bf16_kernel(
    const float* __restrict__ src, unsigned short* __restrict__ dst, size_t n) {
  const size_t i = (size_t)blockIdx.x * 256 + threadIdx.x;
  if (i < n) dst[i] = f2bf(src[i]);
}

__global__ __launch_bounds__(256) void bias_sum_kernel(
    const float* __restrict__ a, const float* __restrict__ b,
    float* __restrict__ dst, int n) {
  const int i = (int)(blockIdx.x * 256 + threadIdx.x);
  if (i < n) dst[i] = a[i] + b[i];
}

__global__ __launch_bounds__(256) void init_state_kernel(
    const float* __restrict__ hxs, const float* __restrict__ cxs,
    unsigned short* __restrict__ hb, float* __restrict__ hf,
    float* __restrict__ cf, int n) {
  const int i = (int)(blockIdx.x * 256 + threadIdx.x);
  if (i < n) { hb[i] = f2bf(hxs[i]); hf[i] = hxs[i]; cf[i] = cxs[i]; }
}

__global__ __launch_bounds__(256) void copy_hn_cn_kernel(
    const float* __restrict__ hf, const float* __restrict__ cf,
    float* __restrict__ hn, float* __restrict__ cn, int n) {
  const int i = (int)(blockIdx.x * 256 + threadIdx.x);
  if (i < n) { hn[i] = hf[i]; cn[i] = cf[i]; }
}

// ---------------------------------------------------------------------------
// Host launch
// ---------------------------------------------------------------------------
extern "C" void kernel_launch(void* const* d_in, const int* in_sizes, int n_in,
                              void* d_out, int out_size, void* d_ws,
                              size_t ws_size, hipStream_t stream) {
  (void)in_sizes; (void)n_in; (void)out_size; (void)ws_size;
  const float* obs = (const float*)d_in[0];
  const float* hxs = (const float*)d_in[1];
  const float* cxs = (const float*)d_in[2];
  const float* W1  = (const float*)d_in[3];
  const float* b1  = (const float*)d_in[4];
  const float* W2  = (const float*)d_in[5];
  const float* b2  = (const float*)d_in[6];
  const float* Wih = (const float*)d_in[7];
  const float* Whh = (const float*)d_in[8];
  const float* bih = (const float*)d_in[9];
  const float* bhh = (const float*)d_in[10];
  const float* lng = (const float*)d_in[11];
  const float* lnb = (const float*)d_in[12];
  const float* Wa  = (const float*)d_in[13];
  const float* ba  = (const float*)d_in[14];
  const float* Wc  = (const float*)d_in[15];
  const float* bc  = (const float*)d_in[16];

  // Workspace carve-out (256B aligned).
  char* ws = (char*)d_ws;
  size_t off = 0;
  auto alloc = [&](size_t bytes) -> char* {
    char* p = ws + off;
    off = (off + bytes + 255) & ~(size_t)255;
    return p;
  };
  unsigned short* obs_b = (unsigned short*)alloc((size_t)Bz * Tz * OBSz * 2);
  unsigned short* W1b   = (unsigned short*)alloc((size_t)ENCz * OBSz * 2);
  unsigned short* W2b   = (unsigned short*)alloc((size_t)ENCz * ENCz * 2);
  unsigned short* Wihb  = (unsigned short*)alloc((size_t)FHz * ENCz * 2);
  unsigned short* Whhb  = (unsigned short*)alloc((size_t)FHz * Hz * 2);
  unsigned short* Wab   = (unsigned short*)alloc((size_t)Az * Hz * 2);
  unsigned short* enc1  = (unsigned short*)alloc((size_t)Bz * Tz * ENCz * 2);
  unsigned short* enc2  = (unsigned short*)alloc((size_t)Bz * Tz * ENCz * 2);
  float*          gx    = (float*)alloc((size_t)Bz * Tz * FHz * 4);
  float*          outf  = (float*)alloc((size_t)Bz * Tz * Hz * 4);
  unsigned short* outb  = (unsigned short*)alloc((size_t)Bz * Tz * Hz * 2);
  unsigned short* hb0   = (unsigned short*)alloc((size_t)Bz * Hz * 2);
  unsigned short* hb1   = (unsigned short*)alloc((size_t)Bz * Hz * 2);
  float*          hf    = (float*)alloc((size_t)Bz * Hz * 4);
  float*          cf    = (float*)alloc((size_t)Bz * Hz * 4);
  float*          bsum  = (float*)alloc((size_t)FHz * 4);
  unsigned int*   ctr   = (unsigned int*)alloc(256);

  // Output layout (flat, return order).
  float* o_logits = (float*)d_out;                        // B*T*A
  float* o_values = o_logits + (size_t)Bz * Tz * Az;      // B*T
  float* o_hn     = o_values + (size_t)Bz * Tz;           // B*H
  float* o_cn     = o_hn + (size_t)Bz * Hz;               // B*H
  float* o_ar     = o_cn + (size_t)Bz * Hz;               // 1
  float* o_tar    = o_ar + 1;                             // 1

  auto cdiv = [](size_t a, size_t b) { return (unsigned)((a + b - 1) / b); };

  // 1) fp32 -> bf16 conversions + bias sum + state init + counter zero.
  f32_to_bf16_kernel<<<cdiv((size_t)Bz * Tz * OBSz, 256), 256, 0, stream>>>(
      obs, obs_b, (size_t)Bz * Tz * OBSz);
  f32_to_bf16_kernel<<<cdiv((size_t)ENCz * OBSz, 256), 256, 0, stream>>>(
      W1, W1b, (size_t)ENCz * OBSz);
  f32_to_bf16_kernel<<<cdiv((size_t)ENCz * ENCz, 256), 256, 0, stream>>>(
      W2, W2b, (size_t)ENCz * ENCz);
  f32_to_bf16_kernel<<<cdiv((size_t)FHz * ENCz, 256), 256, 0, stream>>>(
      Wih, Wihb, (size_t)FHz * ENCz);
  f32_to_bf16_kernel<<<cdiv((size_t)FHz * Hz, 256), 256, 0, stream>>>(
      Whh, Whhb, (size_t)FHz * Hz);
  f32_to_bf16_kernel<<<cdiv((size_t)Az * Hz, 256), 256, 0, stream>>>(
      Wa, Wab, (size_t)Az * Hz);
  bias_sum_kernel<<<cdiv(FHz, 256), 256, 0, stream>>>(bih, bhh, bsum, FHz);
  init_state_kernel<<<cdiv((size_t)Bz * Hz, 256), 256, 0, stream>>>(
      hxs, cxs, hb0, hf, cf, Bz * Hz);
  zero_scalars_kernel<<<1, 64, 0, stream>>>(o_ar, o_tar, ctr);

  const int M = Bz * Tz;  // 16384
  // 2) Encoder GEMM1: (M x ENC) = obs @ W1^T, SiLU -> bf16.
  wmma_gemm_kernel<<<(M / 32) * (ENCz / 64) / 8, 256, 0, stream>>>(
      obs_b, W1b, b1, nullptr, enc1, M, ENCz, OBSz, 1);
  // 3) Encoder GEMM2: SiLU -> bf16.
  wmma_gemm_kernel<<<(M / 32) * (ENCz / 64) / 8, 256, 0, stream>>>(
      enc1, W2b, b2, nullptr, enc2, M, ENCz, ENCz, 1);
  // 4) gx = enc @ W_ih^T + (b_ih + b_hh) -> f32.
  wmma_gemm_kernel<<<(M / 32) * (FHz / 64) / 8, 256, 0, stream>>>(
      enc2, Wihb, bsum, gx, nullptr, M, FHz, ENCz, 0);

  // 5) LSTM recurrence: single persistent cooperative kernel with W_hh in LDS.
  lstm_persistent_kernel<<<LSTM_BLOCKS, 256, LSTM_SMEM_BYTES, stream>>>(
      gx, Whhb, hb0, hb1, cf, outf, hf, ctr);

  // 6) h_n / c_n (pre-LayerNorm hidden state).
  copy_hn_cn_kernel<<<cdiv((size_t)Bz * Hz, 256), 256, 0, stream>>>(
      hf, cf, o_hn, o_cn, Bz * Hz);

  // 7) LayerNorm (in place) + bf16 copy for heads.
  ln_kernel<<<M, 256, 0, stream>>>(outf, outb, lng, lnb);

  // 8) Losses.
  loss_kernel<<<2048, 256, 0, stream>>>(outf, o_ar, o_tar);

  // 9) Actor head logits (WMMA) straight into d_out.
  wmma_gemm_kernel<<<(M / 32) * (Az / 64) / 8, 256, 0, stream>>>(
      outb, Wab, ba, o_logits, nullptr, M, Az, Hz, 0);

  // 10) Value head.
  values_kernel<<<cdiv((size_t)M * 32, 256), 256, 0, stream>>>(
      outf, Wc, bc, o_values);
}
